// GAT_71760313582053
// MI455X (gfx1250) — matrix operand order repaired and checked by reading, not compiled, not run on hardware
//
#include <hip/hip_runtime.h>
#include <hip/hip_bf16.h>

// GAT attention, algebraically fused for MI455X (gfx1250, wave32):
//   sim[i,j] = x_self . (Wq@wa_q) + x[i,j] . (Wk@wa_k) + c0      (GEMV, VALU)
//   out[i]   = (sum_j attn[i,j] * x[i,j]) @ Wv + (sum_j attn)*bv  (WMMA bf16)
// Memory-bound: ~112 MB streamed via global_load_async_to_lds_b128.

#define B_ 256
#define N_ 64
#define KN 12
#define KP1 13
#define D_ 128
#define TN 16            // nodes per workgroup
#define ROWS (TN * KP1)  // 208

typedef __bf16 bf16_t;
typedef bf16_t v16bf __attribute__((ext_vector_type(16)));
typedef float v8f __attribute__((ext_vector_type(8)));
typedef unsigned int uu4 __attribute__((ext_vector_type(4)));

union FragBF { v16bf v; uu4 q[2]; };

__device__ __forceinline__ unsigned short f2bf(float f) {
  unsigned u = __float_as_uint(f);
  unsigned r = u + 0x7FFFu + ((u >> 16) & 1u);  // round-to-nearest-even
  return (unsigned short)(r >> 16);
}

// CDNA5 async global->LDS copy (ASYNCcnt), 16B per lane. vdst = LDS byte
// offset VGPR, vaddr = 64-bit global address VGPR pair.
__device__ __forceinline__ void async_copy_b128(unsigned lds_off,
                                                unsigned long long gaddr) {
  asm volatile("global_load_async_to_lds_b128 %0, %1, off"
               :: "v"(lds_off), "v"(gaddr)
               : "memory");
}

__device__ __forceinline__ void wait_async_all() {
  asm volatile("s_wait_asynccnt 0" ::: "memory");
}

// ---------------------------------------------------------------------------
// Prep kernel: Wv^T as bf16 into d_ws, plus wqv = Wq@wa_q, wkv = Wk@wa_k,
// c0 = bq.wa_q + bk.wa_k + ba.
// ---------------------------------------------------------------------------
__global__ void __launch_bounds__(256) gat_prep(
    const float* __restrict__ Wq, const float* __restrict__ Wk,
    const float* __restrict__ Wv, const float* __restrict__ bq,
    const float* __restrict__ bk, const float* __restrict__ waq,
    const float* __restrict__ wak, const float* __restrict__ ba,
    unsigned short* __restrict__ wvt, float* __restrict__ wqv,
    float* __restrict__ wkv, float* __restrict__ c0) {
  if (blockIdx.x < 64) {
    int idx = blockIdx.x * 256 + threadIdx.x;   // 0..16383
    int n = idx >> 7;
    int k = idx & 127;
    wvt[n * D_ + k] = f2bf(Wv[k * D_ + n]);     // Wv^T[n][k] = Wv[k][n]
  } else {
    int t = threadIdx.x;
    if (t < 128) {
      float s = 0.f;
      for (int n = 0; n < D_; ++n) s = fmaf(Wq[t * D_ + n], waq[n], s);
      wqv[t] = s;
    } else {
      int r = t - 128;
      float s = 0.f;
      for (int n = 0; n < D_; ++n) s = fmaf(Wk[r * D_ + n], wak[n], s);
      wkv[r] = s;
    }
    if (t == 0) {
      float s = ba[0];
      for (int n = 0; n < D_; ++n) s = fmaf(bq[n], waq[n], s);
      for (int n = 0; n < D_; ++n) s = fmaf(bk[n], wak[n], s);
      c0[0] = s;
    }
  }
}

// ---------------------------------------------------------------------------
// Main fused kernel: one workgroup per (batch, 16-node tile).
// ---------------------------------------------------------------------------
__global__ void __launch_bounds__(256) gat_fused(
    const float* __restrict__ nodes, const float* __restrict__ neigh,
    const float* __restrict__ mask, const float* __restrict__ bv,
    const unsigned short* __restrict__ wvt_g, const float* __restrict__ wqv_g,
    const float* __restrict__ wkv_g, const float* __restrict__ c0_g,
    float* __restrict__ out) {
  __shared__ __align__(16) float nh[TN][KP1][D_];         // 106496 B, fp32 rows
  __shared__ __align__(16) unsigned short wvt[D_][D_];    // 32768 B, Wv^T bf16
  __shared__ __align__(16) unsigned short xbar[TN][D_];   // 4096 B, bf16
  __shared__ __align__(16) float attn[ROWS];
  __shared__ __align__(16) float svals[TN];
  __shared__ __align__(16) float wqv_l[D_];
  __shared__ __align__(16) float wkv_l[D_];

  const int t = threadIdx.x;
  const int wave = t >> 5;
  const int lane = t & 31;
  const int b = blockIdx.x >> 2;
  const int n0 = (blockIdx.x & 3) << 4;

  // small vectors: plain path
  if (t < D_) wqv_l[t] = wqv_g[t];
  else        wkv_l[t - D_] = wkv_g[t - D_];

  // ---- async bulk staging into LDS (TDM-style async path, ASYNCcnt) ----
  {
    unsigned wlds = (unsigned)(unsigned long long)(const void*)&wvt[0][0];
    unsigned long long wsrc = (unsigned long long)wvt_g;
    #pragma unroll
    for (int it = 0; it < 8; ++it) {               // 32 KB of Wv^T
      unsigned off = (unsigned)((t + (it << 8)) << 4);
      async_copy_b128(wlds + off, wsrc + off);
    }
    unsigned nlds = (unsigned)(unsigned long long)(const void*)&nh[0][0][0];
    for (int rr = 0; rr < 26; ++rr) {              // 208 rows, 512 B each
      int r = wave + (rr << 3);
      int i = r / KP1;
      int j = r - i * KP1;
      const float* src = (j == 0)
          ? (nodes + (((size_t)(b * N_ + n0 + i)) << 7))
          : (neigh + ((((size_t)(b * N_ + n0 + i)) * KN + (j - 1)) << 7));
      async_copy_b128(nlds + (unsigned)(r * 512 + (lane << 4)),
                      (unsigned long long)src + (unsigned)(lane << 4));
    }
  }
  wait_async_all();
  __syncthreads();

  // ---- similarity -> LeakyReLU -> exp -> mask ----
  if (t < ROWS) {
    const int i = t / KP1;
    const int j = t - i * KP1;
    const float* xs = &nh[i][0][0];
    const float* xr = &nh[i][j][0];
    float sq = 0.f, sk = 0.f;
    #pragma unroll 4
    for (int d = 0; d < D_; ++d) {
      sq = fmaf(xs[d], wqv_l[d], sq);
      sk = fmaf(xr[d], wkv_l[d], sk);
    }
    float sim = sq + sk + c0_g[0];
    sim = (sim >= 0.f) ? sim : 0.2f * sim;
    float m = (j == 0) ? 1.f
                       : mask[((size_t)(b * N_ + n0 + i)) * KN + (j - 1)];
    attn[t] = __expf(sim) * m;
  }
  __syncthreads();

  // ---- normalize; s_i = sum(attn) (exact bv coefficient) ----
  if (t < TN) {
    float s = 0.f;
    #pragma unroll
    for (int j = 0; j < KP1; ++j) s += attn[t * KP1 + j];
    float inv = 1.f / (s + 1e-16f);
    #pragma unroll
    for (int j = 0; j < KP1; ++j) attn[t * KP1 + j] *= inv;
    svals[t] = s * inv;
  }
  __syncthreads();

  // ---- xbar[i] = sum_j attn[i,j] * x[i,j,:]  (fp32 accum, bf16 store) ----
  {
    const int i = t >> 4;
    const int dbase = (t & 15) << 3;
    float acc[8];
    #pragma unroll
    for (int dd = 0; dd < 8; ++dd) acc[dd] = 0.f;
    #pragma unroll
    for (int j = 0; j < KP1; ++j) {
      const float a = attn[i * KP1 + j];
      const float* xr = &nh[i][j][dbase];
      #pragma unroll
      for (int dd = 0; dd < 8; ++dd) acc[dd] = fmaf(a, xr[dd], acc[dd]);
    }
    uu4 pk;
    pk.x = (unsigned)f2bf(acc[0]) | ((unsigned)f2bf(acc[1]) << 16);
    pk.y = (unsigned)f2bf(acc[2]) | ((unsigned)f2bf(acc[3]) << 16);
    pk.z = (unsigned)f2bf(acc[4]) | ((unsigned)f2bf(acc[5]) << 16);
    pk.w = (unsigned)f2bf(acc[6]) | ((unsigned)f2bf(acc[7]) << 16);
    *(uu4*)&xbar[i][dbase] = pk;
  }
  __syncthreads();

  // ---- out tile = xbar @ Wv + s*bv via v_wmma_f32_16x16x32_bf16 ----
  // Each wave owns one 16-column tile; 4 k-steps of 32.
  {
    const int cbase = wave << 4;
    const int nn = lane & 15;        // A row (M) and B/D column (N)
    const int khalf = lane >> 4;     // lane half selects K sub-range / D rows
    const float bvv = bv[cbase + nn];
    v8f acc;
    #pragma unroll
    for (int r = 0; r < 8; ++r) acc[r] = svals[r + (khalf << 3)] * bvv;
    #pragma unroll
    for (int kt = 0; kt < 4; ++kt) {
      const int kk = kt << 5;
      FragBF a, bm;
      // A 16x32 bf16: lanes 0-15 K=kk+0..7 & kk+16..23; lanes 16-31 +8
      a.q[0] = *(const uu4*)&xbar[nn][kk + (khalf << 3)];
      a.q[1] = *(const uu4*)&xbar[nn][kk + 16 + (khalf << 3)];
      // B 32x16 bf16: lane = column; lanes 0-15 K=kk+0..7 & kk+16..23; lanes 16-31 +8
      bm.q[0] = *(const uu4*)&wvt[cbase + nn][kk + (khalf << 3)];
      bm.q[1] = *(const uu4*)&wvt[cbase + nn][kk + 16 + (khalf << 3)];
      acc = __builtin_amdgcn_wmma_f32_16x16x32_bf16(
          false, a.v, false, bm.v, (short)0, acc, false, false);
    }
    float* op = out + (((size_t)(b * N_ + n0)) << 7) + cbase + nn;
    #pragma unroll
    for (int r = 0; r < 8; ++r) {
      const int m2 = r + (khalf << 3);   // D layout: VGPR r -> M=r / r+8
      op[(size_t)m2 << 7] = acc[r];
    }
  }
}

extern "C" void kernel_launch(void* const* d_in, const int* in_sizes, int n_in,
                              void* d_out, int out_size, void* d_ws,
                              size_t ws_size, hipStream_t stream) {
  (void)in_sizes; (void)n_in; (void)out_size; (void)ws_size;
  const float* nodes = (const float*)d_in[0];
  const float* neigh = (const float*)d_in[1];
  const float* mask  = (const float*)d_in[2];
  const float* Wq    = (const float*)d_in[3];
  const float* bq    = (const float*)d_in[4];
  const float* Wk    = (const float*)d_in[5];
  const float* bk    = (const float*)d_in[6];
  const float* Wv    = (const float*)d_in[7];
  const float* bv    = (const float*)d_in[8];
  const float* waq   = (const float*)d_in[9];
  const float* wak   = (const float*)d_in[10];
  const float* ba    = (const float*)d_in[11];

  unsigned short* wvt = (unsigned short*)d_ws;                 // 32768 B
  float* wqv = (float*)((char*)d_ws + 32768);                  // 512 B
  float* wkv = wqv + 128;                                      // 512 B
  float* c0  = wkv + 128;                                      // 4 B

  gat_prep<<<65, 256, 0, stream>>>(Wq, Wk, Wv, bq, bk, waq, wak, ba,
                                   wvt, wqv, wkv, c0);
  gat_fused<<<B_ * (N_ / TN), 256, 0, stream>>>(nodes, neigh, mask, bv, wvt,
                                                wqv, wkv, c0, (float*)d_out);
}
